// Block_54116587929701
// MI455X (gfx1250) — matrix operand order repaired and checked by reading, not compiled, not run on hardware
//
#include <hip/hip_runtime.h>
#include <math.h>

// ---------------- problem constants ----------------
#define EMB    768
#define HEADS  12
#define HDIM   64
#define HIDDEN 3072
#define SEQ    2048
#define BATCH  4
#define ROWS   (BATCH * SEQ)   // 8192 token rows

// ---------------- WMMA types (gfx1250, wave32) ----------------
typedef __bf16        v16bf  __attribute__((ext_vector_type(16)));
typedef float         v8f    __attribute__((ext_vector_type(8)));
typedef unsigned short u16x8 __attribute__((ext_vector_type(8)));

union Frag16 { v16bf v; u16x8 h[2]; };

__device__ __forceinline__ unsigned short f2bf(float f) {
  unsigned int u = __float_as_uint(f);
  u += 0x7fffu + ((u >> 16) & 1u);      // round-to-nearest-even
  return (unsigned short)(u >> 16);
}

__device__ __forceinline__ unsigned int lds_off(const void* p) {
  // addrspace(3) offset == low 32 bits of the generic pointer
  return (unsigned int)(size_t)p;
}

// one 16B async global->LDS copy (GVS mode: sgpr base + vgpr byte offset)
__device__ __forceinline__ void async_b128(unsigned int lds_byte,
                                           unsigned int goff_byte,
                                           const void* sbase) {
  asm volatile("global_load_async_to_lds_b128 %0, %1, %2"
               :: "v"(lds_byte), "v"(goff_byte), "s"(sbase) : "memory");
}

__device__ __forceinline__ void wait_async_le(int n) {
  if (n == 0)       asm volatile("s_wait_asynccnt 0x0" ::: "memory");
  else if (n == 2)  asm volatile("s_wait_asynccnt 0x2" ::: "memory");
  else              asm volatile("s_wait_asynccnt 0x4" ::: "memory");
}

// ---------------- fp32 -> bf16, transposed (W[k][n] -> WT[n][k]) ----------------
__global__ void f32_to_bf16_T(const float* __restrict__ in,
                              unsigned short* __restrict__ out, int K, int N) {
  int i = blockIdx.x * blockDim.x + threadIdx.x;
  if (i < K * N) {
    int k = i / N, n = i % N;
    out[(size_t)n * K + k] = f2bf(in[i]);
  }
}

// ---------------- fused LayerNorm -> bf16 ----------------
__global__ __launch_bounds__(256)
void layernorm_bf16(const float* __restrict__ x, const float* __restrict__ g,
                    const float* __restrict__ b, unsigned short* __restrict__ out) {
  __shared__ float red[256];
  const int row = blockIdx.x, tid = threadIdx.x;
  const float* xr = x + (size_t)row * EMB;
  float s = 0.f;
  for (int i = tid; i < EMB; i += 256) s += xr[i];
  red[tid] = s; __syncthreads();
  for (int st = 128; st > 0; st >>= 1) { if (tid < st) red[tid] += red[tid + st]; __syncthreads(); }
  const float mu = red[0] * (1.0f / EMB);
  __syncthreads();
  float v = 0.f;
  for (int i = tid; i < EMB; i += 256) { float d = xr[i] - mu; v += d * d; }
  red[tid] = v; __syncthreads();
  for (int st = 128; st > 0; st >>= 1) { if (tid < st) red[tid] += red[tid + st]; __syncthreads(); }
  const float rstd = rsqrtf(red[0] * (1.0f / EMB) + 1e-5f);
  for (int i = tid; i < EMB; i += 256)
    out[(size_t)row * EMB + i] = f2bf((xr[i] - mu) * rstd * g[i] + b[i]);
}

// ---------------- bf16 WMMA GEMM, async double-buffered LDS ----------------
// C = A(MxK, row-major) * WT(NxK, row-major)^T  — i.e. classic A*B with B=WT^T.
// EPI 0: QKV scatter + bias -> bf16 q/k [b,h,m,d], v [b,h,d,m]
// EPI 1: bias + residual(f32) -> f32
// EPI 2: bias + exact GELU -> bf16
template <int EPI>
__global__ __launch_bounds__(256)
void gemm_bf16(const unsigned short* __restrict__ A,    // M x K bf16
               const unsigned short* __restrict__ WT,   // N x K bf16 (pre-transposed)
               const float* __restrict__ bias,
               const float* __restrict__ resid,
               float* __restrict__ outF,
               unsigned short* __restrict__ outB,
               unsigned short* __restrict__ outQ,
               unsigned short* __restrict__ outK,
               unsigned short* __restrict__ outV,
               int M, int N, int K) {
  constexpr int KC = 64;
  __shared__ __align__(16) unsigned short lA[2][64 * KC];   // [row][k]
  __shared__ __align__(16) unsigned short lB[2][64 * KC];   // [n][k]
  const int tid  = threadIdx.x;
  const int w    = tid >> 5, lane = tid & 31;
  const int half = lane >> 4, lr = lane & 15;
  const int wm   = w & 3, wn = w >> 2;            // 4 waves in M, 2 in N
  const int m0   = blockIdx.y * 64, n0 = blockIdx.x * 64;

  const unsigned short* Abase = A  + (size_t)m0 * K;
  const unsigned short* Bbase = WT + (size_t)n0 * K;
  // per-thread slice of the 64x64 tile: 512 x 16B ops, 2 per thread
  const int trow = tid >> 2, tseg = tid & 3;          // row 0..63, seg 0..3 (x8 halfwords)
  const int trow2 = trow + 32;                        // second op (rows 32..63 shifted): use +64? no:
  // ops: idx = tid and tid+256 -> rows tid>>2 and (tid+256)>>2 = trow+64? 256>>2=64 -> out of range.
  // Use idx mapping: row = idx >> 3, seg8 = idx & 7 over 64 rows x 8 segs? KC=64 halfwords = 128B = 8x16B.
  // 64 rows * 8 segs = 512 ops. op0: row = tid>>3? only 32 rows. Do: i in {0,1}: idx = tid + i*256.
  (void)trow; (void)tseg; (void)trow2;

  auto prefetch = [&](int t) {
#pragma unroll
    for (int i = 0; i < 2; ++i) {
      int idx = tid + i * 256;            // 0..511
      int row = idx >> 3, seg = idx & 7;  // 64 rows, 8 x 16B segments (KC=64 halfwords)
      unsigned int lo = (unsigned int)((row * KC + seg * 8) * 2);
      unsigned int go = (unsigned int)(((size_t)row * K + t * KC + seg * 8) * 2);
      async_b128(lds_off(&lA[t & 1][0]) + lo, go, Abase);
      async_b128(lds_off(&lB[t & 1][0]) + lo, go, Bbase);
    }
  };

  v8f acc0 = {0.f,0.f,0.f,0.f,0.f,0.f,0.f,0.f};
  v8f acc1 = {0.f,0.f,0.f,0.f,0.f,0.f,0.f,0.f};

  const int nt = K / KC;
  prefetch(0);
  for (int t = 0; t < nt; ++t) {
    if (t + 1 < nt) prefetch(t + 1);
    wait_async_le(t + 1 < nt ? 4 : 0);    // our 4 ops for tile t have landed
    __syncthreads();                      // everyone's ops for tile t have landed
    const unsigned short* tA = &lA[t & 1][0];
    const unsigned short* tB = &lB[t & 1][0];
#pragma unroll
    for (int c = 0; c < KC / 32; ++c) {
      Frag16 a;
      const int ab = (wm * 16 + lr) * KC + c * 32;       // A: row lr, K(e,half)=e+16*half
      a.h[0] = *(const u16x8*)&tA[ab + 16 * half];
      a.h[1] = *(const u16x8*)&tA[ab + 16 * half + 8];
      Frag16 b0, b1;                                      // B: 16 contiguous K per lane-half
      const int bb0 = (wn * 32 + lr) * KC + c * 32 + 16 * half;
      b0.h[0] = *(const u16x8*)&tB[bb0];
      b0.h[1] = *(const u16x8*)&tB[bb0 + 8];
      const int bb1 = (wn * 32 + 16 + lr) * KC + c * 32 + 16 * half;
      b1.h[0] = *(const u16x8*)&tB[bb1];
      b1.h[1] = *(const u16x8*)&tB[bb1 + 8];
      acc0 = __builtin_amdgcn_wmma_f32_16x16x32_bf16(false, a.v, false, b0.v,
                                                     (short)0, acc0, false, false);
      acc1 = __builtin_amdgcn_wmma_f32_16x16x32_bf16(false, a.v, false, b1.v,
                                                     (short)0, acc1, false, false);
    }
    __syncthreads();                      // tile consumed; safe to overwrite next iter
  }

  // epilogue: C/D layout M = v + 8*half, N = lr
#pragma unroll
  for (int ntl = 0; ntl < 2; ++ntl) {
    const int col = n0 + wn * 32 + ntl * 16 + lr;
    const float bv = bias[col];
#pragma unroll
    for (int vv = 0; vv < 8; ++vv) {
      const int gm = m0 + wm * 16 + 8 * half + vv;
      float val = (ntl ? acc1[vv] : acc0[vv]) + bv;
      if (EPI == 0) {
        int o = col % 3, head = col / 192, dd = (col / 3) % 64;
        int bb = gm >> 11, pos = gm & 2047;
        unsigned short hv = f2bf(val);
        if (o == 0)
          outQ[((size_t)(bb * HEADS + head) * SEQ + pos) * HDIM + dd] = hv;
        else if (o == 1)
          outK[((size_t)(bb * HEADS + head) * SEQ + pos) * HDIM + dd] = hv;
        else  // V stored transposed [b,h,d,m] so flash V-tiles are straight copies
          outV[((size_t)(bb * HEADS + head) * HDIM + dd) * SEQ + pos] = hv;
      } else if (EPI == 1) {
        size_t di = (size_t)gm * N + col;
        outF[di] = val + resid[di];
      } else {
        float ge = 0.5f * val * (1.0f + erff(val * 0.70710678118654752f));
        outB[(size_t)gm * N + col] = f2bf(ge);
      }
    }
  }
}

// ---------------- flash attention (streaming softmax, WMMA, async tiles) ----------------
__global__ __launch_bounds__(256)
void flash_attn(const unsigned short* __restrict__ Q,    // [b,h,m,d]
                const unsigned short* __restrict__ Kb,   // [b,h,m,d]
                const unsigned short* __restrict__ Vt,   // [b,h,d,m]  (pre-transposed)
                unsigned short* __restrict__ O) {        // [b,m,emb]
  __shared__ __align__(16) unsigned short lK[2][32 * 64];   // [key][d]
  __shared__ __align__(16) unsigned short lV[2][64 * 32];   // [d][key]
  __shared__ __align__(16) unsigned short lP[8 * 16 * 32];  // per-wave P staging
  const int tid  = threadIdx.x;
  const int w    = tid >> 5, lane = tid & 31;
  const int half = lane >> 4, lr = lane & 15;
  const int qblk = blockIdx.x, head = blockIdx.y, bat = blockIdx.z;
  const size_t hbase = (size_t)(bat * HEADS + head) * SEQ * HDIM;
  const int qr0 = qblk * 128 + w * 16;                      // this wave's 16 q rows

  const unsigned short* Kbase = Kb + hbase;
  const unsigned short* Vbase = Vt + hbase;                 // [d][m] for this (b,h)

  auto prefetch = [&](int j) {
    // K tile: 32 keys x 64 hw = 32 rows x 8 segs = 256 ops, 1/thread
    {
      int key = tid >> 3, seg = tid & 7;
      unsigned int lo = (unsigned int)((key * 64 + seg * 8) * 2);
      unsigned int go = (unsigned int)(((size_t)(j * 32 + key) * HDIM + seg * 8) * 2);
      async_b128(lds_off(&lK[j & 1][0]) + lo, go, Kbase);
    }
    // V tile: 64 d x 32 keys = 64 rows x 4 segs = 256 ops, 1/thread
    {
      int d = tid >> 2, seg = tid & 3;
      unsigned int lo = (unsigned int)((d * 32 + seg * 8) * 2);
      unsigned int go = (unsigned int)(((size_t)d * SEQ + j * 32 + seg * 8) * 2);
      async_b128(lds_off(&lV[j & 1][0]) + lo, go, Vbase);
    }
  };

  // Q A-fragments, loaded once (d = K-dim, 2 chunks of 32)
  Frag16 qa[2];
  const unsigned short* qp = Q + hbase + (size_t)(qr0 + lr) * HDIM;
#pragma unroll
  for (int c = 0; c < 2; ++c) {
    qa[c].h[0] = *(const u16x8*)(qp + c * 32 + 16 * half);
    qa[c].h[1] = *(const u16x8*)(qp + c * 32 + 16 * half + 8);
  }

  const float inv_scale = 0.0360843918243516f;              // 1/sqrt(768)
  float mrow[8], lsum[8];
  v8f oacc[4];
#pragma unroll
  for (int vv = 0; vv < 8; ++vv) { mrow[vv] = -1e30f; lsum[vv] = 0.f; }
#pragma unroll
  for (int t = 0; t < 4; ++t)
#pragma unroll
    for (int vv = 0; vv < 8; ++vv) oacc[t][vv] = 0.f;

  const int NJ = SEQ / 32;
  prefetch(0);
  for (int j = 0; j < NJ; ++j) {
    if (j + 1 < NJ) prefetch(j + 1);
    wait_async_le(j + 1 < NJ ? 2 : 0);
    __syncthreads();
    const unsigned short* tK = &lK[j & 1][0];
    const unsigned short* tV = &lV[j & 1][0];

    // S = Q K^T : 16 rows x 32 keys, K-dim = 64 (2 chunks)
    v8f s0 = {0.f,0.f,0.f,0.f,0.f,0.f,0.f,0.f};
    v8f s1 = {0.f,0.f,0.f,0.f,0.f,0.f,0.f,0.f};
#pragma unroll
    for (int c = 0; c < 2; ++c) {
      Frag16 b0, b1;                       // B[kd][key] = K[key][kd]
      const int kb0 = lr * 64 + c * 32 + 16 * half;
      b0.h[0] = *(const u16x8*)&tK[kb0];
      b0.h[1] = *(const u16x8*)&tK[kb0 + 8];
      const int kb1 = (16 + lr) * 64 + c * 32 + 16 * half;
      b1.h[0] = *(const u16x8*)&tK[kb1];
      b1.h[1] = *(const u16x8*)&tK[kb1 + 8];
      s0 = __builtin_amdgcn_wmma_f32_16x16x32_bf16(false, qa[c].v, false, b0.v,
                                                   (short)0, s0, false, false);
      s1 = __builtin_amdgcn_wmma_f32_16x16x32_bf16(false, qa[c].v, false, b1.v,
                                                   (short)0, s1, false, false);
    }

    // online softmax (row r = 8*half+vv lives in this lane's 16-lane half)
    float p0[8], p1[8], scl[8];
#pragma unroll
    for (int vv = 0; vv < 8; ++vv) {
      float a = s0[vv] * inv_scale, b = s1[vv] * inv_scale;
      float mx = fmaxf(a, b);
#pragma unroll
      for (int off = 8; off > 0; off >>= 1) mx = fmaxf(mx, __shfl_xor(mx, off, 16));
      float nm = fmaxf(mrow[vv], mx);
      float sc = __expf(mrow[vv] - nm);
      float e0 = __expf(a - nm), e1 = __expf(b - nm);
      float sm = e0 + e1;
#pragma unroll
      for (int off = 8; off > 0; off >>= 1) sm += __shfl_xor(sm, off, 16);
      lsum[vv] = lsum[vv] * sc + sm;
      mrow[vv] = nm;
      p0[vv] = e0; p1[vv] = e1; scl[vv] = sc;
    }
#pragma unroll
    for (int t = 0; t < 4; ++t)
#pragma unroll
      for (int vv = 0; vv < 8; ++vv) oacc[t][vv] *= scl[vv];

    // stage P (C-layout -> A-layout) through this wave's private LDS patch
    const int wp = w * 512;
#pragma unroll
    for (int vv = 0; vv < 8; ++vv) {
      int r = 8 * half + vv;
      lP[wp + r * 32 + lr]      = f2bf(p0[vv]);
      lP[wp + r * 32 + 16 + lr] = f2bf(p1[vv]);
    }
    asm volatile("s_wait_dscnt 0x0" ::: "memory");   // per-wave DScnt: our stores landed
    Frag16 pa;
    pa.h[0] = *(const u16x8*)&lP[wp + lr * 32 + 16 * half];
    pa.h[1] = *(const u16x8*)&lP[wp + lr * 32 + 16 * half + 8];

    // O += P V : B[key][d] read from [d][key] tile
#pragma unroll
    for (int t = 0; t < 4; ++t) {
      Frag16 bv;
      const int vb0 = (t * 16 + lr) * 32 + 16 * half;
      bv.h[0] = *(const u16x8*)&tV[vb0];
      bv.h[1] = *(const u16x8*)&tV[vb0 + 8];
      oacc[t] = __builtin_amdgcn_wmma_f32_16x16x32_bf16(false, pa.v, false, bv.v,
                                                        (short)0, oacc[t], false, false);
    }
    __syncthreads();
  }

  // normalize and write back to [b, m, emb] bf16
#pragma unroll
  for (int t = 0; t < 4; ++t) {
    const int d = t * 16 + lr;
#pragma unroll
    for (int vv = 0; vv < 8; ++vv) {
      const int r = 8 * half + vv;
      const int pos = qr0 + r;
      float val = oacc[t][vv] / lsum[vv];
      O[((size_t)bat * SEQ + pos) * EMB + head * HDIM + d] = f2bf(val);
    }
  }
}

// ---------------- host orchestration ----------------
extern "C" void kernel_launch(void* const* d_in, const int* in_sizes, int n_in,
                              void* d_out, int out_size, void* d_ws, size_t ws_size,
                              hipStream_t stream) {
  (void)in_sizes; (void)n_in; (void)out_size; (void)ws_size;
  const float* x      = (const float*)d_in[0];
  const float* ln1_g  = (const float*)d_in[1];
  const float* ln1_b  = (const float*)d_in[2];
  const float* w_qkv  = (const float*)d_in[3];
  const float* b_qkv  = (const float*)d_in[4];
  const float* w_proj = (const float*)d_in[5];
  const float* b_proj = (const float*)d_in[6];
  const float* ln2_g  = (const float*)d_in[7];
  const float* ln2_b  = (const float*)d_in[8];
  const float* w_fc1  = (const float*)d_in[9];
  const float* b_fc1  = (const float*)d_in[10];
  const float* w_fc2  = (const float*)d_in[11];
  const float* b_fc2  = (const float*)d_in[12];
  float* out = (float*)d_out;

  // workspace carve-up (256B aligned slices)
  char* ws = (char*)d_ws;
  size_t cur = 0;
  auto take = [&](size_t bytes) -> char* {
    char* p = ws + cur;
    cur += (bytes + 255) & ~(size_t)255;
    return p;
  };
  unsigned short* wqkvT = (unsigned short*)take((size_t)EMB * 3 * EMB * 2);  // [3E][E]
  unsigned short* wprojT= (unsigned short*)take((size_t)EMB * EMB * 2);      // [E][E]
  unsigned short* wfc1T = (unsigned short*)take((size_t)EMB * HIDDEN * 2);   // [H][E]
  unsigned short* wfc2T = (unsigned short*)take((size_t)HIDDEN * EMB * 2);   // [E][H]
  unsigned short* h1_bf   = (unsigned short*)take((size_t)ROWS * EMB * 2);
  unsigned short* qb      = (unsigned short*)take((size_t)ROWS * EMB * 2);
  unsigned short* kb      = (unsigned short*)take((size_t)ROWS * EMB * 2);
  unsigned short* vbT     = (unsigned short*)take((size_t)ROWS * EMB * 2);   // [b,h,d,m]
  unsigned short* attn_bf = (unsigned short*)take((size_t)ROWS * EMB * 2);
  float*          x1      = (float*)take((size_t)ROWS * EMB * 4);
  unsigned short* h2_bf   = (unsigned short*)take((size_t)ROWS * EMB * 2);
  unsigned short* fc1_bf  = (unsigned short*)take((size_t)ROWS * HIDDEN * 2);

  // 1) weights -> bf16, transposed to [N][K]
  {
    int n;
    n = EMB * 3 * EMB;  f32_to_bf16_T<<<(n + 255) / 256, 256, 0, stream>>>(w_qkv, wqkvT, EMB, 3 * EMB);
    n = EMB * EMB;      f32_to_bf16_T<<<(n + 255) / 256, 256, 0, stream>>>(w_proj, wprojT, EMB, EMB);
    n = EMB * HIDDEN;   f32_to_bf16_T<<<(n + 255) / 256, 256, 0, stream>>>(w_fc1, wfc1T, EMB, HIDDEN);
    n = HIDDEN * EMB;   f32_to_bf16_T<<<(n + 255) / 256, 256, 0, stream>>>(w_fc2, wfc2T, HIDDEN, EMB);
  }
  // 2) LN1
  layernorm_bf16<<<ROWS, 256, 0, stream>>>(x, ln1_g, ln1_b, h1_bf);
  // 3) QKV GEMM + scatter  (8192 x 2304 x 768)
  gemm_bf16<0><<<dim3(3 * EMB / 64, ROWS / 64), 256, 0, stream>>>(
      h1_bf, wqkvT, b_qkv, nullptr, nullptr, nullptr, qb, kb, vbT,
      ROWS, 3 * EMB, EMB);
  // 4) flash attention
  flash_attn<<<dim3(SEQ / 128, HEADS, BATCH), 256, 0, stream>>>(qb, kb, vbT, attn_bf);
  // 5) proj + residual -> x1 (f32)
  gemm_bf16<1><<<dim3(EMB / 64, ROWS / 64), 256, 0, stream>>>(
      attn_bf, wprojT, b_proj, x, x1, nullptr, nullptr, nullptr, nullptr,
      ROWS, EMB, EMB);
  // 6) LN2
  layernorm_bf16<<<ROWS, 256, 0, stream>>>(x1, ln2_g, ln2_b, h2_bf);
  // 7) FC1 + GELU -> bf16
  gemm_bf16<2><<<dim3(HIDDEN / 64, ROWS / 64), 256, 0, stream>>>(
      h2_bf, wfc1T, b_fc1, nullptr, nullptr, fc1_bf, nullptr, nullptr, nullptr,
      ROWS, HIDDEN, EMB);
  // 8) FC2 + residual -> final f32 output
  gemm_bf16<1><<<dim3(EMB / 64, ROWS / 64), 256, 0, stream>>>(
      fc1_bf, wfc2T, b_fc2, x1, out, nullptr, nullptr, nullptr, nullptr,
      ROWS, EMB, HIDDEN);
}
